// MainModel_80118319940151
// MI455X (gfx1250) — matrix operand order repaired and checked by reading, not compiled, not run on hardware
//
#include <hip/hip_runtime.h>
#include <hip/hip_bf16.h>

// ---------------------------------------------------------------------------
// Problem constants (match reference)
// ---------------------------------------------------------------------------
#define NNODES 100000
#define NEDGES 1600000
#define DIM    128
#define EPS_LN 1e-12f
#define EPS_BN 1e-5f

// ---------------------------------------------------------------------------
// CDNA5 WMMA vector types (gfx1250, wave32)
// ---------------------------------------------------------------------------
typedef __attribute__((ext_vector_type(16))) __bf16 v16bf;
typedef __attribute__((ext_vector_type(8)))  __bf16 v8bf;
typedef __attribute__((ext_vector_type(8)))  float  v8f;

// fp32 -> bf16 round-to-nearest-even
__device__ __forceinline__ __bf16 f2bf(float f) {
    unsigned int u = __float_as_uint(f);
    unsigned short r = (unsigned short)((u + 0x7FFFu + ((u >> 16) & 1u)) >> 16);
    __bf16 out;
    __builtin_memcpy(&out, &r, 2);
    return out;
}

// ---------------------------------------------------------------------------
// Kernel: fill float buffer with a constant
// ---------------------------------------------------------------------------
__global__ void k_fill(float* __restrict__ p, float v, int n) {
    int i = blockIdx.x * blockDim.x + threadIdx.x;
    if (i < n) p[i] = v;
}

// ---------------------------------------------------------------------------
// Kernel: embedding gather (4 tables) + LayerNorm.  One wave per node,
// lane handles 4 columns (d = lane + 32*j).  Wave32 shfl reductions.
// Writes fp32 x and bf16 mirror for the WMMA GEMM.
// ---------------------------------------------------------------------------
__global__ void k_embed_ln(const int* __restrict__ ids,
                           const float* __restrict__ syn,
                           const float* __restrict__ pos,
                           const float* __restrict__ sen,
                           const float* __restrict__ lem,
                           const float* __restrict__ g,
                           const float* __restrict__ b,
                           float* __restrict__ x,
                           __bf16* __restrict__ xbf,
                           int nnodes) {
    int wave = (blockIdx.x * blockDim.x + threadIdx.x) >> 5;
    int lane = threadIdx.x & 31;
    if (wave >= nnodes) return;

    int i0 = ids[wave * 4 + 0];   // synset
    int i1 = ids[wave * 4 + 1];   // pos
    int i2 = ids[wave * 4 + 2];   // sense
    int i3 = ids[wave * 4 + 3];   // lemma

    float v[4];
    float s = 0.f;
#pragma unroll
    for (int j = 0; j < 4; ++j) {
        int d = lane + 32 * j;
        float e = syn[i0 * DIM + d] + lem[i3 * DIM + d]
                + pos[i1 * DIM + d] + sen[i2 * DIM + d];
        v[j] = e;
        s += e;
    }
#pragma unroll
    for (int off = 16; off; off >>= 1) s += __shfl_xor(s, off, 32);
    float mu = s * (1.f / DIM);

    float vs = 0.f;
#pragma unroll
    for (int j = 0; j < 4; ++j) { float d = v[j] - mu; vs += d * d; }
#pragma unroll
    for (int off = 16; off; off >>= 1) vs += __shfl_xor(vs, off, 32);
    float rinv = rsqrtf(vs * (1.f / DIM) + EPS_LN);

#pragma unroll
    for (int j = 0; j < 4; ++j) {
        int d = lane + 32 * j;
        float o = (v[j] - mu) * rinv * g[d] + b[d];
        x[wave * DIM + d] = o;
        xbf[wave * DIM + d] = f2bf(o);
    }
}

// ---------------------------------------------------------------------------
// Kernel: degree count (deg pre-initialized to 1.0 for self loop)
// ---------------------------------------------------------------------------
__global__ void k_deg_count(const int* __restrict__ dst, float* __restrict__ deg, int ne) {
    int e = blockIdx.x * blockDim.x + threadIdx.x;
    if (e < ne) atomicAdd(&deg[dst[e]], 1.0f);
}

__global__ void k_dinv(float* __restrict__ deg, int n) {
    int i = blockIdx.x * blockDim.x + threadIdx.x;
    if (i < n) deg[i] = rsqrtf(deg[i]);   // in-place: deg -> deg^{-1/2}
}

// ---------------------------------------------------------------------------
// Kernel: convert W (fp32 [K=128][N=128], row-major) -> TRANSPOSED bf16
//   wt[n*D + k] = bf16(W[k*D + n])
// so that a B fragment (16 consecutive K values of one column n) is one
// 32-byte contiguous run.
// ---------------------------------------------------------------------------
__global__ void k_cvt_wt(const float* __restrict__ w, __bf16* __restrict__ wt, int d) {
    int i = blockIdx.x * blockDim.x + threadIdx.x;   // i = k*D + n (coalesced read)
    if (i < d * d) {
        int k = i >> 7;          // / DIM
        int n = i & (DIM - 1);
        wt[(size_t)n * DIM + k] = f2bf(w[i]);
    }
}

// ---------------------------------------------------------------------------
// Kernel: h = x @ W via v_wmma_f32_16x16x32_bf16.
//
// The whole transposed-W bf16 matrix (32 KB) is staged into LDS once per
// workgroup (WGP has 320 KB), so every B fragment is a pair of low-latency
// ds_load_b128 instead of an L2 round-trip, and the block's 8 waves share
// one copy (8x less L2 traffic).
//
// One wave per 16-row tile (6250 tiles, N divides exactly).  Wave computes
// all 8 column tiles, K loop of 4 steps of 32.  A fragments preloaded once
// (global b128 pairs) and reused across the 8 column tiles.
//
// Fragment layouts used here (wave32; A and B share one contiguous K order):
//   A (16x32 bf16):  lane m = lane&15; lane half selects K offset +0/+16;
//                    elems 0-7 -> K = khalf+0..7, elems 8-15 -> K = khalf+8..15
//   B (32x16 bf16):  column n = lane&15; lanes 0-15 hold K 0-15,
//                    lanes 16-31 hold K 16-31; elem e -> K = koff+e
//                    => with Wt row-major one 32-byte contiguous run per lane.
//   C/D (16x16 f32): VGPR r -> row = r + 8*(lane>=16), col = lane&15.
// ---------------------------------------------------------------------------
__global__ void k_gemm_wmma(const __bf16* __restrict__ xbf,
                            const __bf16* __restrict__ wt,
                            float* __restrict__ h,
                            int ntiles) {
    __shared__ __bf16 lw[DIM * DIM];     // 32 KB: whole transposed W

    // Cooperative stage: 256 threads x 8 x 16 B = 32 KB (b128 copies)
    {
        const float4* srcp = reinterpret_cast<const float4*>(wt);
        float4*       dstp = reinterpret_cast<float4*>(lw);
#pragma unroll
        for (int it = 0; it < (DIM * DIM / 8) / 256; ++it)
            dstp[it * 256 + threadIdx.x] = srcp[it * 256 + threadIdx.x];
    }
    __syncthreads();

    int wave = (blockIdx.x * blockDim.x + threadIdx.x) >> 5;
    int lane = threadIdx.x & 31;
    if (wave >= ntiles) return;          // uniform per wave: EXEC stays all-1s

    int mt    = wave;
    int m     = lane & 15;
    int row   = mt * 16 + m;
    int upper = lane >> 4;               // 0 for lanes 0-15, 1 for lanes 16-31
    int khalf = upper ? 16 : 0;

    // Preload A fragments for the 4 K-steps (reused across all 8 col tiles)
    v16bf afr[4];
#pragma unroll
    for (int kt = 0; kt < 4; ++kt) {
        int kb = kt * 32;
        const v8bf* p0 = reinterpret_cast<const v8bf*>(xbf + (size_t)row * DIM + kb + khalf);
        const v8bf* p1 = reinterpret_cast<const v8bf*>(xbf + (size_t)row * DIM + kb + khalf + 8);
        v8bf lo = *p0;
        v8bf hi = *p1;
        afr[kt] = __builtin_shufflevector(lo, hi,
                                          0, 1, 2, 3, 4, 5, 6, 7,
                                          8, 9, 10, 11, 12, 13, 14, 15);
    }

    int n    = lane & 15;
    int koff = upper ? 16 : 0;
    int orow = mt * 16 + (upper ? 8 : 0);

#pragma unroll
    for (int nt = 0; nt < 8; ++nt) {
        // B fragments: 32-byte contiguous LDS reads (ds_load_b128 pairs)
        const __bf16* wrow = lw + (size_t)(nt * 16 + n) * DIM + koff;
        v8f acc = {};
#pragma unroll
        for (int kt = 0; kt < 4; ++kt) {
            v16bf bfr = *reinterpret_cast<const v16bf*>(wrow + kt * 32);
            acc = __builtin_amdgcn_wmma_f32_16x16x32_bf16(
                      false, afr[kt], false, bfr, (short)0, acc, false, false);
        }
        float* hp = h + (size_t)orow * DIM + nt * 16 + n;
#pragma unroll
        for (int r = 0; r < 8; ++r) hp[(size_t)r * DIM] = acc[r];
    }
}

// ---------------------------------------------------------------------------
// Kernel: edge scatter  agg[dst] += h[src] * dinv[src]*dinv[dst]
// One wave per edge; lane handles 4 consecutive floats (float4 row read,
// 512 B per row fully covered by the wave), per-element f32 atomics.
// ---------------------------------------------------------------------------
__global__ void k_scatter(const int* __restrict__ src,
                          const int* __restrict__ dst,
                          const float* __restrict__ dinv,
                          const float* __restrict__ h,
                          float* __restrict__ agg,
                          int ne) {
    int wave = (blockIdx.x * blockDim.x + threadIdx.x) >> 5;
    int lane = threadIdx.x & 31;
    if (wave >= ne) return;
    int s = src[wave];
    int d = dst[wave];
    float norm = dinv[s] * dinv[d];
    const float4 hv = reinterpret_cast<const float4*>(h + (size_t)s * DIM)[lane];
    float* ap = agg + (size_t)d * DIM + lane * 4;
    atomicAdd(ap + 0, hv.x * norm);
    atomicAdd(ap + 1, hv.y * norm);
    atomicAdd(ap + 2, hv.z * norm);
    atomicAdd(ap + 3, hv.w * norm);
}

// ---------------------------------------------------------------------------
// Kernel: layer finalize: x = relu(agg + h*dinv^2 + bias); writes fp32 + bf16,
// and accumulates per-column pooling partials (max via int atomics — values
// are >= 0 after ReLU so int bit-compare is order-correct; sum via f32 add).
// Block = 128 threads (one per column), each block covers FIN_ROWS rows.
// ---------------------------------------------------------------------------
#define FIN_ROWS 256
__global__ void k_finalize(const float* __restrict__ agg,
                           const float* __restrict__ h,
                           const float* __restrict__ dinv,
                           const float* __restrict__ bias,
                           float* __restrict__ x,
                           __bf16* __restrict__ xbf,
                           int* __restrict__ poolmax,
                           float* __restrict__ poolsum,
                           int nnodes) {
    int d = threadIdx.x;                 // 0..127
    int row0 = blockIdx.x * FIN_ROWS;
    float bd = bias[d];
    float lmax = 0.f, lsum = 0.f;
    for (int r = 0; r < FIN_ROWS; ++r) {
        int i = row0 + r;
        if (i >= nnodes) break;
        float sn = dinv[i];
        sn *= sn;
        float v = agg[(size_t)i * DIM + d] + h[(size_t)i * DIM + d] * sn + bd;
        v = fmaxf(v, 0.f);
        x[(size_t)i * DIM + d] = v;
        xbf[(size_t)i * DIM + d] = f2bf(v);
        lmax = fmaxf(lmax, v);
        lsum += v;
    }
    atomicMax(poolmax + d, __float_as_int(lmax));
    atomicAdd(poolsum + d, lsum);
}

// ---------------------------------------------------------------------------
// Kernel: accumulate graph embedding, reset pool buffers. 1 block, 128 thr.
// ---------------------------------------------------------------------------
__global__ void k_pool_accum(float* __restrict__ gacc,
                             int* __restrict__ poolmax,
                             float* __restrict__ poolsum,
                             float invN) {
    int d = threadIdx.x;
    gacc[d]       += __int_as_float(poolmax[d]);
    gacc[DIM + d] += poolsum[d] * invN;
    poolmax[d] = 0;
    poolsum[d] = 0.f;
}

// ---------------------------------------------------------------------------
// Kernel: BatchNorm1d (eval) -> node_emb in d_out[0 : N*D)
// ---------------------------------------------------------------------------
__global__ void k_bn(const float* __restrict__ x,
                     const float* __restrict__ g,
                     const float* __restrict__ b,
                     const float* __restrict__ mean,
                     const float* __restrict__ var,
                     float* __restrict__ out,
                     int total) {
    int i = blockIdx.x * blockDim.x + threadIdx.x;
    if (i >= total) return;
    int d = i & (DIM - 1);
    out[i] = (x[i] - mean[d]) * rsqrtf(var[d] + EPS_BN) * g[d] + b[d];
}

// ---------------------------------------------------------------------------
// Kernel: write graph_emb (256 floats) after node_emb
// ---------------------------------------------------------------------------
__global__ void k_write_graph(const float* __restrict__ gacc,
                              float* __restrict__ out) {
    int j = threadIdx.x;
    if (j < 2 * DIM) out[j] = gacc[j];
}

// ---------------------------------------------------------------------------
// Host launcher
// ---------------------------------------------------------------------------
extern "C" void kernel_launch(void* const* d_in, const int* in_sizes, int n_in,
                              void* d_out, int out_size, void* d_ws, size_t ws_size,
                              hipStream_t stream) {
    const int N = in_sizes[0] / 4;   // 100000
    const int E = in_sizes[1] / 2;   // 1600000
    const int D = DIM;

    const int*   x_ids   = (const int*)d_in[0];
    const int*   e_src   = (const int*)d_in[1];
    const int*   e_dst   = e_src + E;
    const float* syn_emb = (const float*)d_in[2];
    const float* pos_emb = (const float*)d_in[3];
    const float* sen_emb = (const float*)d_in[4];
    const float* lem_emb = (const float*)d_in[5];
    const float* ln_g    = (const float*)d_in[6];
    const float* ln_b    = (const float*)d_in[7];
    const float* W[3]    = {(const float*)d_in[8],  (const float*)d_in[10], (const float*)d_in[12]};
    const float* B[3]    = {(const float*)d_in[9],  (const float*)d_in[11], (const float*)d_in[13]};
    const float* bn_g    = (const float*)d_in[14];
    const float* bn_b    = (const float*)d_in[15];
    const float* bn_mean = (const float*)d_in[16];
    const float* bn_var  = (const float*)d_in[17];

    // ---- workspace carve-out (256-byte aligned slices) ----
    char* ws = (char*)d_ws;
    size_t off = 0;
    auto carve = [&](size_t bytes) -> char* {
        char* p = ws + off;
        off += (bytes + 255) & ~(size_t)255;
        return p;
    };
    float*  x_buf  = (float*)carve((size_t)N * D * sizeof(float));
    float*  h_buf  = (float*)carve((size_t)N * D * sizeof(float));
    float*  a_buf  = (float*)carve((size_t)N * D * sizeof(float));
    __bf16* xbf    = (__bf16*)carve((size_t)N * D * sizeof(__bf16));
    __bf16* wt     = (__bf16*)carve((size_t)D * D * sizeof(__bf16));
    float*  dinv   = (float*)carve((size_t)N * sizeof(float));
    int*    pmax   = (int*)carve(D * sizeof(int));
    float*  psum   = (float*)carve(D * sizeof(float));
    float*  gacc   = (float*)carve(2 * D * sizeof(float));
    (void)ws_size;

    const int ND = N * D;

    // ---- init: deg=1 (self-loop), pools/gacc = 0 ----
    k_fill<<<(N + 255) / 256, 256, 0, stream>>>(dinv, 1.0f, N);         // deg
    k_fill<<<1, 256, 0, stream>>>((float*)pmax, 0.0f, D);               // int 0 == 0.0f
    k_fill<<<1, 256, 0, stream>>>(psum, 0.0f, D);
    k_fill<<<1, 256, 0, stream>>>(gacc, 0.0f, 2 * D);

    // ---- degree + deg^{-1/2} ----
    k_deg_count<<<(E + 255) / 256, 256, 0, stream>>>(e_dst, dinv, E);
    k_dinv<<<(N + 255) / 256, 256, 0, stream>>>(dinv, N);

    // ---- embeddings + LayerNorm ----
    {
        int waves = N, blocks = (waves * 32 + 255) / 256;
        k_embed_ln<<<blocks, 256, 0, stream>>>(x_ids, syn_emb, pos_emb, sen_emb,
                                               lem_emb, ln_g, ln_b, x_buf, xbf, N);
    }

    // ---- 3 GCN layers ----
    const int ntiles = N / 16;                         // 6250, exact
    const int gemm_blocks = (ntiles * 32 + 255) / 256; // 8 waves / block
    for (int l = 0; l < 3; ++l) {
        k_cvt_wt<<<(D * D + 255) / 256, 256, 0, stream>>>(W[l], wt, D);
        k_gemm_wmma<<<gemm_blocks, 256, 0, stream>>>(xbf, wt, h_buf, ntiles);
        k_fill<<<(ND + 255) / 256, 256, 0, stream>>>(a_buf, 0.0f, ND);
        {
            int blocks = (E * 32 + 255) / 256;         // wave per edge
            k_scatter<<<blocks, 256, 0, stream>>>(e_src, e_dst, dinv, h_buf, a_buf, E);
        }
        k_finalize<<<(N + FIN_ROWS - 1) / FIN_ROWS, D, 0, stream>>>(
            a_buf, h_buf, dinv, B[l], x_buf, xbf, pmax, psum, N);
        k_pool_accum<<<1, D, 0, stream>>>(gacc, pmax, psum, 1.0f / (float)N);
    }

    // ---- outputs: node_emb (BN) then graph_emb ----
    float* out = (float*)d_out;
    k_bn<<<(ND + 255) / 256, 256, 0, stream>>>(x_buf, bn_g, bn_b, bn_mean, bn_var, out, ND);
    k_write_graph<<<1, 256, 0, stream>>>(gacc, out + (size_t)ND);
    (void)out_size; (void)n_in;
}
